// MultiHeadAttention_38036230373429
// MI455X (gfx1250) — matrix operand order repaired and checked by reading, not compiled, not run on hardware
//
#include <hip/hip_runtime.h>
#include <hip/hip_bf16.h>

// ---------------------------------------------------------------------------
// MHA for MI455X (gfx1250), wave32. All GEMMs via v_wmma_f32_16x16x32_f16.
// Fragment-order layouts -> every WMMA operand is 2x b128 loads.
// Flash-attention online softmax; async global->LDS staging when available.
// ---------------------------------------------------------------------------

typedef __attribute__((ext_vector_type(16))) _Float16 v16h;
typedef __attribute__((ext_vector_type(8)))  _Float16 v8h;
typedef __attribute__((ext_vector_type(4)))  _Float16 v4h;
typedef __attribute__((ext_vector_type(8)))  float    v8f;

constexpr int D_MODEL = 1024;
constexpr int SEQ     = 2048;
constexpr int BATCH   = 4;
constexpr int HEADS   = 16;
constexpr int DKH     = 64;
constexpr int ROWS    = BATCH * SEQ;                  // 8192 tokens
constexpr size_t MAT  = (size_t)ROWS * D_MODEL;       // 8M elements
constexpr size_t QSTR = (size_t)SEQ * DKH;            // per-bh Q block

#if defined(__gfx1250__) && \
    __has_builtin(__builtin_amdgcn_global_load_async_to_lds_b128) && \
    __has_builtin(__builtin_amdgcn_s_wait_asynccnt)
#define USE_ASYNC_LDS 1
// Exact pointee type from the builtin's prototype (vector_size, not ext_vector).
typedef int b128i __attribute__((vector_size(16)));
typedef __attribute__((address_space(1))) b128i* gas1_b128;
typedef __attribute__((address_space(3))) b128i* las3_b128;
#else
#define USE_ASYNC_LDS 0
#endif

// ---- WMMA wrapper ----------------------------------------------------------
__device__ __forceinline__ v8f wmma_f16(v16h a, v16h b, v8f c) {
  return __builtin_amdgcn_wmma_f32_16x16x32_f16(false, a, false, b,
                                                (short)0, c, false, false);
}

// ---- A-fragment swizzle (ISA 7.12.2, 16-bit A 16x32) -----------------------
__device__ __forceinline__ int a_lane(int kd, int m) {
  return (((kd >> 3) & 1) << 4) + m;
}
__device__ __forceinline__ int a_half(int kd) {
  const int v = ((kd & 16) >> 2) + ((kd & 7) >> 1);
  return v * 2 + (kd & 1);
}

// ---- contiguous 16-half fragment load (16B-aligned) ------------------------
__device__ __forceinline__ v16h frag_load(const _Float16* p) {
  const v8h lo = *(const v8h*)(p);
  const v8h hi = *(const v8h*)(p + 8);
  return __builtin_shufflevector(lo, hi, 0, 1, 2, 3, 4, 5, 6, 7,
                                 8, 9, 10, 11, 12, 13, 14, 15);
}
__device__ __forceinline__ v16h load_afrag(const _Float16* base, int lane_stride) {
  const int lane = threadIdx.x & 31;
  return frag_load(base + (size_t)lane * lane_stride);
}
// B fragment: row-per-N storage, halves h<8 -> K = (lane/16)*8+h, h>=8 -> K = 16+(lane/16)*8+(h-8).
__device__ __forceinline__ v16h load_bfrag(const _Float16* base, int n0, int ld) {
  const int lane = threadIdx.x & 31;
  const _Float16* p = base + (size_t)(n0 + (lane & 15)) * ld + ((lane >> 4) << 3);
  const v8h lo = *(const v8h*)(p);
  const v8h hi = *(const v8h*)(p + 16);
  return __builtin_shufflevector(lo, hi, 0, 1, 2, 3, 4, 5, 6, 7,
                                 8, 9, 10, 11, 12, 13, 14, 15);
}

// ---------------------------------------------------------------------------
// Kernel 1: QKV projection. grid=(16, 64, 3), block=256 (8 waves, 128x64 tile)
//  Q -> ws in A-fragment order:  [bh][strip(128)][lane(32)][chunk(2)*16]
//  K -> ws row-major:            [bh][s][64]
//  V -> ws transposed:           [bh][d(64)][s(2048)]
// ---------------------------------------------------------------------------
__global__ __launch_bounds__(256, 1)
void qkv_proj_kernel(const float* __restrict__ X,
                     const float* __restrict__ Wq,
                     const float* __restrict__ Wk,
                     const float* __restrict__ Wv,
                     _Float16* __restrict__ qws,
                     _Float16* __restrict__ kws,
                     _Float16* __restrict__ vws) {
  __shared__ _Float16 sXf[8 * 528];   // A tile, fragment order, 8 strips
  __shared__ _Float16 sWf[64 * 40];   // B tile, column-strip (16B rows)

  const int tid  = threadIdx.x;
  const int wave = tid >> 5;
  const int lane = tid & 31;
  const int row0 = blockIdx.y * 128;
  const int col0 = blockIdx.x * 64;
  const int z    = blockIdx.z;
  const float* W = (z == 0) ? Wq : ((z == 1) ? Wk : Wv);

  // W staging ownership: column n = tid%64, k-octet kq = tid/64 (4 octets).
  const int wn = tid & 63, wk = (tid >> 6) * 8;

  v8f acc[4] = {};

  for (int k0 = 0; k0 < D_MODEL; k0 += 32) {
    // ---- stage X tile (128x32) pre-swizzled: float4 load -> v4h store ----
#pragma unroll
    for (int it = 0; it < 4; ++it) {
      const int i = tid + it * 256;                  // 1024 quads
      const int r = i >> 3;                          // row 0..127
      const int k = (i & 7) * 4;                     // k 0,4,..,28
      const float4 xv = *(const float4*)(X + (size_t)(row0 + r) * D_MODEL + k0 + k);
      v4h hp;
      hp[0] = (_Float16)xv.x; hp[1] = (_Float16)xv.y;
      hp[2] = (_Float16)xv.z; hp[3] = (_Float16)xv.w;
      const int strip = r >> 4, m = r & 15;
      *(v4h*)(&sXf[strip * 528 + a_lane(k, m) * 16 + a_half(k)]) = hp;
    }
    // ---- stage W tile (32x64): 8 coalesced dword loads -> one b128 store ----
    {
      v8h wv;
#pragma unroll
      for (int j = 0; j < 8; ++j)
        wv[j] = (_Float16)W[(size_t)(k0 + wk + j) * D_MODEL + col0 + wn];
      *(v8h*)(&sWf[wn * 40 + wk]) = wv;
    }
    __syncthreads();

    const v16h a = load_afrag(&sXf[wave * 528], 16);
#pragma unroll
    for (int t = 0; t < 4; ++t) {
      const v16h b = load_bfrag(sWf, t * 16, 40);
      acc[t] = wmma_f16(a, b, acc[t]);
    }
    __syncthreads();
  }

  // ---- scatter results to Q/K/V staging layouts ----
  const int rh = lane >> 4, nc = lane & 15;
#pragma unroll
  for (int t = 0; t < 4; ++t) {
#pragma unroll
    for (int r = 0; r < 8; ++r) {
      const int grow = row0 + wave * 16 + rh * 8 + r;      // token 0..8191
      const int gcol = col0 + t * 16 + nc;                 // 0..1023
      const int b = grow >> 11, s = grow & (SEQ - 1);
      const int h = gcol >> 6, d = gcol & (DKH - 1);
      const int bh = b * HEADS + h;
      const _Float16 val = (_Float16)acc[t][r];
      if (z == 0) {
        const int strip = s >> 4, m = s & 15;
        const int ch = d >> 5, kd = d & 31;
        qws[(size_t)bh * QSTR + (size_t)strip * 1024 +
            a_lane(kd, m) * 32 + ch * 16 + a_half(kd)] = val;
      } else if (z == 1) {
        kws[((size_t)bh * SEQ + s) * DKH + d] = val;
      } else {
        vws[((size_t)bh * DKH + d) * SEQ + s] = val;
      }
    }
  }
}

// ---------------------------------------------------------------------------
// Kernel 2: causal flash attention. grid=(SEQ/128, B*H), block=256 (8 waves).
// 128 query rows per WG; Q fragments in registers; K/V tiles in LDS
// (async global->LDS when the builtins exist). Attended out in A-frag order.
// ---------------------------------------------------------------------------
__global__ __launch_bounds__(256, 1)
void attn_kernel(const _Float16* __restrict__ q_ws,
                 const _Float16* __restrict__ k_ws,
                 const _Float16* __restrict__ v_ws,
                 _Float16* __restrict__ att_ws) {
  __shared__ _Float16 sK[64][72];        // [key][d]   (B^T frags contiguous)
  __shared__ _Float16 sVt[64][72];       // [d][key]   (B frags contiguous)
  __shared__ _Float16 sPf[8 * 2 * 528];  // P in A-frag order per wave/chunk

  const int tid  = threadIdx.x;
  const int wave = tid >> 5;
  const int lane = tid & 31;
  const int bh   = blockIdx.y;
  const int q0   = blockIdx.x * 128;
  const int rh   = lane >> 4, nc = lane & 15;
  const float NEG_INF = -__builtin_inff();

  // This wave's 16-row Q strip: 2 K-chunks in registers for the whole loop.
  const _Float16* qbase =
      q_ws + (size_t)bh * QSTR + (size_t)((q0 >> 4) + wave) * 1024;
  const v16h qf0 = load_afrag(qbase, 32);
  const v16h qf1 = load_afrag(qbase + 16, 32);

  float m_run[8], l_run[8];
#pragma unroll
  for (int r = 0; r < 8; ++r) { m_run[r] = NEG_INF; l_run[r] = 0.0f; }
  v8f Oacc[4] = {};

  for (int j0 = 0; j0 <= q0 + 64; j0 += 64) {
    const size_t kb = ((size_t)bh * SEQ + j0) * DKH;
    const size_t vb = (size_t)bh * DKH * SEQ + j0;
#if USE_ASYNC_LDS
#pragma unroll
    for (int it = 0; it < 2; ++it) {
      const int i = tid + it * 256;                 // 512 octets
      const int r = i >> 3, c8 = (i & 7) * 8;
      __builtin_amdgcn_global_load_async_to_lds_b128(
          (gas1_b128)(k_ws + kb + (size_t)r * DKH + c8),
          (las3_b128)&sK[r][c8], 0, 0);
      __builtin_amdgcn_global_load_async_to_lds_b128(
          (gas1_b128)(v_ws + vb + (size_t)r * SEQ + c8),
          (las3_b128)&sVt[r][c8], 0, 0);
    }
    __builtin_amdgcn_s_wait_asynccnt(0);
#else
#pragma unroll
    for (int it = 0; it < 2; ++it) {
      const int i = tid + it * 256;
      const int r = i >> 3, c8 = (i & 7) * 8;
      *(v8h*)&sK[r][c8]  = *(const v8h*)(k_ws + kb + (size_t)r * DKH + c8);
      *(v8h*)&sVt[r][c8] = *(const v8h*)(v_ws + vb + (size_t)r * SEQ + c8);
    }
#endif
    if (j0 <= q0) {  // prefetch next K/V block
      __builtin_prefetch(k_ws + kb + (size_t)64 * DKH, 0, 1);
      __builtin_prefetch(v_ws + vb + 64, 0, 1);
    }
    __syncthreads();

    // Wave-uniform skip of fully masked key blocks (EXEC stays all-ones).
    const bool active = (j0 <= q0 + wave * 16 + 15);
    v8f Sacc[4] = {};
    if (active) {
      // ---- S = Q * K^T : 16 rows x 64 keys per wave ----
#pragma unroll
      for (int t = 0; t < 4; ++t) {
        Sacc[t] = wmma_f16(qf0, load_bfrag(&sK[0][0], t * 16, 72), Sacc[t]);
        Sacc[t] = wmma_f16(qf1, load_bfrag(&sK[0][32], t * 16, 72), Sacc[t]);
      }
      // ---- scale + causal mask + online softmax per row ----
#pragma unroll
      for (int r = 0; r < 8; ++r) {
        const int qrow = q0 + wave * 16 + rh * 8 + r;
        float bm = NEG_INF;
#pragma unroll
        for (int t = 0; t < 4; ++t) {
          float s = Sacc[t][r] * 0.125f;             // 1/sqrt(64)
          if (j0 + t * 16 + nc > qrow) s = NEG_INF;
          Sacc[t][r] = s;
          bm = fmaxf(bm, s);
        }
#pragma unroll
        for (int off = 1; off < 16; off <<= 1)
          bm = fmaxf(bm, __shfl_xor(bm, off, 32));
        const float mnew  = fmaxf(m_run[r], bm);
        const float alpha = __expf(m_run[r] - mnew);
        float rsum = 0.0f;
#pragma unroll
        for (int t = 0; t < 4; ++t) {
          const float p = __expf(Sacc[t][r] - mnew);
          Sacc[t][r] = p;
          rsum += p;
        }
#pragma unroll
        for (int off = 1; off < 16; off <<= 1)
          rsum += __shfl_xor(rsum, off, 32);
        l_run[r] = l_run[r] * alpha + rsum;
        m_run[r] = mnew;
#pragma unroll
        for (int t = 0; t < 4; ++t) Oacc[t][r] *= alpha;
      }
      // ---- stage P in A-fragment order ----
#pragma unroll
      for (int t = 0; t < 4; ++t)
#pragma unroll
        for (int r = 0; r < 8; ++r) {
          const int key = t * 16 + nc;
          const int ch = key >> 5, kd = key & 31;
          sPf[(wave * 2 + ch) * 528 + a_lane(kd, rh * 8 + r) * 16 + a_half(kd)]
              = (_Float16)Sacc[t][r];
        }
    }
    __syncthreads();   // order P writes before fragment reads

    if (active) {
      const v16h p0 = load_afrag(&sPf[(wave * 2 + 0) * 528], 16);
      const v16h p1 = load_afrag(&sPf[(wave * 2 + 1) * 528], 16);
#pragma unroll
      for (int t = 0; t < 4; ++t) {
        Oacc[t] = wmma_f16(p0, load_bfrag(&sVt[0][0], t * 16, 72), Oacc[t]);
        Oacc[t] = wmma_f16(p1, load_bfrag(&sVt[0][32], t * 16, 72), Oacc[t]);
      }
    }
    __syncthreads();   // protect sK/sVt/sPf before next iteration
  }

  // ---- normalize; store attended in A-fragment order for out_proj ----
  const int b = bh >> 4, h = bh & (HEADS - 1);
#pragma unroll
  for (int r = 0; r < 8; ++r) {
    const float inv = 1.0f / l_run[r];
    const int grow = b * SEQ + q0 + wave * 16 + rh * 8 + r;   // token id
    const int strip = grow >> 4, m = grow & 15;
#pragma unroll
    for (int t = 0; t < 4; ++t) {
      const int gcol = h * DKH + t * 16 + nc;                 // 0..1023
      const int ch = gcol >> 5, kd = gcol & 31;
      att_ws[((size_t)strip * 32 + ch) * 512 + a_lane(kd, m) * 16 + a_half(kd)]
          = (_Float16)(Oacc[t][r] * inv);
    }
  }
}

// ---------------------------------------------------------------------------
// Kernel 3: out = attended(f16, A-frag order) @ W_o -> f32.
// grid=(16, 64), block=256. A frags straight from global; LDS only for W.
// ---------------------------------------------------------------------------
__global__ __launch_bounds__(256, 1)
void out_proj_kernel(const _Float16* __restrict__ aws,
                     const float* __restrict__ Wo,
                     float* __restrict__ out) {
  __shared__ _Float16 sWf[64 * 40];

  const int tid  = threadIdx.x;
  const int wave = tid >> 5;
  const int lane = tid & 31;
  const int row0 = blockIdx.y * 128;
  const int col0 = blockIdx.x * 64;
  const int strip = (row0 >> 4) + wave;
  const int wn = tid & 63, wk = (tid >> 6) * 8;

  v8f acc[4] = {};

  for (int k0 = 0; k0 < D_MODEL; k0 += 32) {
    {
      v8h wv;
#pragma unroll
      for (int j = 0; j < 8; ++j)
        wv[j] = (_Float16)Wo[(size_t)(k0 + wk + j) * D_MODEL + col0 + wn];
      *(v8h*)(&sWf[wn * 40 + wk]) = wv;
    }
    __syncthreads();

    const v16h a =
        load_afrag(aws + ((size_t)strip * 32 + (k0 >> 5)) * 512, 16);
#pragma unroll
    for (int t = 0; t < 4; ++t) {
      const v16h b = load_bfrag(sWf, t * 16, 40);
      acc[t] = wmma_f16(a, b, acc[t]);
    }
    __syncthreads();
  }

  const int rh = lane >> 4, nc = lane & 15;
#pragma unroll
  for (int t = 0; t < 4; ++t)
#pragma unroll
    for (int r = 0; r < 8; ++r) {
      const int grow = row0 + wave * 16 + rh * 8 + r;
      const int gcol = col0 + t * 16 + nc;
      out[(size_t)grow * D_MODEL + gcol] = acc[t][r];
    }
}

// ---------------------------------------------------------------------------
extern "C" void kernel_launch(void* const* d_in, const int* in_sizes, int n_in,
                              void* d_out, int out_size, void* d_ws, size_t ws_size,
                              hipStream_t stream) {
  const float* X  = (const float*)d_in[0];
  const float* Wq = (const float*)d_in[1];
  const float* Wk = (const float*)d_in[2];
  const float* Wv = (const float*)d_in[3];
  const float* Wo = (const float*)d_in[4];

  _Float16* qws = (_Float16*)d_ws;      // A-fragment order per bh
  _Float16* kws = qws + MAT;            // [bh][s][64]
  _Float16* vws = qws + 2 * MAT;        // [bh][d][s]
  _Float16* aws = qws + 3 * MAT;        // attended, A-fragment order

  dim3 blk(256);
  dim3 g1(D_MODEL / 64, ROWS / 128, 3);
  qkv_proj_kernel<<<g1, blk, 0, stream>>>(X, Wq, Wk, Wv, qws, kws, vws);

  dim3 g2(SEQ / 128, BATCH * HEADS);
  attn_kernel<<<g2, blk, 0, stream>>>(qws, kws, vws, aws);

  dim3 g3(D_MODEL / 64, ROWS / 128);
  out_proj_kernel<<<g3, blk, 0, stream>>>(aws, Wo, (float*)d_out);
}
